// CNNDecoder_6811818132334
// MI455X (gfx1250) — matrix operand order repaired and checked by reading, not compiled, not run on hardware
//
#include <hip/hip_runtime.h>
#include <hip/hip_bf16.h>
#include <math.h>

typedef __attribute__((ext_vector_type(16))) _Float16 v16h;
typedef __attribute__((ext_vector_type(8)))  _Float16 v8h;
typedef __attribute__((ext_vector_type(8)))  float    v8f;

// ---------------------------------------------------------------------------
// CDNA5 async global->LDS copy (ASYNCcnt-tracked) + waits, via inline asm
// (portable across ROCm7.2 / amdgpu-toolchain per the bridge doc).
// VDST = 32-bit LDS byte address (low 32 bits of the generic pointer),
// VADDR = 64-bit global address.
// ---------------------------------------------------------------------------
__device__ __forceinline__ void async_copy_b128(const _Float16* g, _Float16* l)
{
    unsigned lofs = (unsigned)(unsigned long long)l;
    asm volatile("global_load_async_to_lds_b128 %0, %1, off"
                 :: "v"(lofs), "v"(g) : "memory");
}
__device__ __forceinline__ void async_wait_2()
{
    asm volatile("s_wait_asynccnt 2" ::: "memory");
}
__device__ __forceinline__ void async_wait_0()
{
    asm volatile("s_wait_asynccnt 0" ::: "memory");
}

// ============================================================================
// Generic batched WMMA GEMM:
//   C[z][M,N] = alpha * A[z][M,K] * Bt[z][N,K]^T  (+ bias[m]) (+ beta*C)
// A row-major [M,K] (lda elems); Bt stores B transposed: row n holds K
// contiguous elements (ldb elems).  Output f32 (Cf) or f16 (Ch, beta ignored).
// tstore=1 writes C[col*ldc + row] (used to emit NHWC / transposed outputs).
// Requirements: N%16==0, K%32==0, pointers 16B aligned, lda/ldb multiples of 8.
// One wave (32 lanes) computes one 16x16 tile; 4 waves / block.
// ============================================================================
__global__ void __launch_bounds__(128)
wmma_gemm_kernel(const _Float16* __restrict__ A, long lda, long sA,
                 const _Float16* __restrict__ Bt, long ldb, long sB,
                 float* __restrict__ Cf, _Float16* __restrict__ Ch, long ldc, long sC,
                 const float* __restrict__ bias,
                 int M, int N, int K, float alpha, float beta, int tstore)
{
    int lane = threadIdx.x;
    int tm   = blockIdx.y * 4 + threadIdx.y;
    if (tm * 16 >= M) return;                 // uniform per wave: EXEC stays full
    int m0 = tm * 16;
    int n0 = blockIdx.x * 16;
    int z  = blockIdx.z;
    A  += (size_t)z * sA;
    Bt += (size_t)z * sB;
    size_t coff = (size_t)z * sC;

    int g  = (lane >> 4) & 1;                 // lane half-group
    int ml = lane & 15;                       // A-row within tile / B-col within tile
    const _Float16* arow = A  + (size_t)(m0 + ml) * lda;
    const _Float16* brow = Bt + (size_t)(n0 + ml) * ldb;

    v8f acc = {};
    for (int k0 = 0; k0 < K; k0 += 32) {
        // A 16x32 f16 fragment: halves 0..7 -> K = g*8..g*8+7 ; 8..15 -> +16
        v8h a0 = *(const v8h*)(arow + k0 + g * 8);
        v8h a1 = *(const v8h*)(arow + k0 + 16 + g * 8);
        // B 32x16 f16 fragment: lane column n, halves 0..7 -> K = g*8..g*8+7 ; 8..15 -> +16
        v8h b0 = *(const v8h*)(brow + k0 + g * 8);
        v8h b1 = *(const v8h*)(brow + k0 + 16 + g * 8);
        v16h af, bf;
#pragma unroll
        for (int i = 0; i < 8; ++i) {
            af[i] = a0[i]; af[i + 8] = a1[i];
            bf[i] = b0[i]; bf[i + 8] = b1[i];
        }
        acc = __builtin_amdgcn_wmma_f32_16x16x32_f16(false, af, false, bf,
                                                     (short)0, acc, false, false);
    }

    int col = n0 + ml;
#pragma unroll
    for (int v = 0; v < 8; ++v) {
        int row = m0 + v + 8 * g;             // C layout: VGPR v, lanes 16-31 -> +8
        float val = alpha * acc[v];
        if (bias) val += bias[row];
        size_t idx = coff + (tstore ? ((size_t)col * ldc + row)
                                    : ((size_t)row * ldc + col));
        if (Ch) {
            Ch[idx] = (_Float16)val;
        } else {
            float prev = (beta != 0.0f) ? Cf[idx] : 0.0f;
            Cf[idx] = val + beta * prev;
        }
    }
}

// ============================================================================
// 3x3 SAME conv as 9-tap shift-GEMM over WMMA, with async global->LDS staging.
//   Xh : NHWC f16 [B,H,W,Cin]      (Cin % 32 == 0)
//   Wp : packed f16 [Cout, 9, Cin] (tap t = kh*3+kw, offset (kh-1, kw-1))
//   Y  : NHWC f32 [B,H,W,Cout], Yh optional NHWC f16 mirror
// Block = 4 waves splitting 64 Cout over ONE 16-pixel tile: the B-operand tile
// (16 pixels x 32 halves) is shared by all 4 waves, so wave 0 stages it into
// LDS with global_load_async_to_lds_b128 (double-buffered, ASYNCcnt-paced,
// overlapped with the WMMAs); consumers read it back with ds_load_b128 from
// rows padded to 40 halves (conflict-free).  Border taps are zero-masked per
// lane at fragment build (data select, EXEC stays all-ones at every WMMA).
// Grid must be exact: (B*H*W)/16 x Cout/64 (true for all shapes used here).
// ============================================================================
__global__ void __launch_bounds__(128)
conv3x3_wmma_kernel(const _Float16* __restrict__ Xh, const _Float16* __restrict__ Wp,
                    const float* __restrict__ bias,
                    float* __restrict__ Y, _Float16* __restrict__ Yh,
                    int H, int W, int Cin, int Cout)
{
    const int LROW = 40;                       // padded row stride (halves)
    __shared__ _Float16 lb[2][16 * LROW];

    int lane = threadIdx.x;
    int wv   = threadIdx.y;
    int co0  = (blockIdx.y * 4 + wv) * 16;
    int n0   = blockIdx.x * 16;
    int g    = (lane >> 4) & 1;
    int ml   = lane & 15;
    int HWi  = H * W;
    int kch  = Cin >> 5;                       // K chunks of 32 per tap
    int L    = 9 * kch;

    // consumer pixel (this lane's B column)
    int j   = n0 + ml;
    int cb  = j / HWi;
    int cr  = j - cb * HWi;
    int chy = cr / W;
    int cwx = cr - chy * W;

    // producer pixel (wave 0; 2 lanes per pixel, 8 halves per async op)
    int pp  = n0 + (lane >> 1);
    int pb  = pp / HWi;
    int pr  = pp - pb * HWi;
    int phy = pr / W;
    int pwx = pr - phy * W;
    int sub = (lane & 1) * 8;

    size_t lda = (size_t)9 * Cin;
    const _Float16* arow = Wp + (size_t)(co0 + ml) * lda;

    auto stage = [&](int buf, int idx) {
        int t  = idx / kch;
        int k0 = (idx - t * kch) << 5;
        int sy = phy + t / 3 - 1;
        int sx = pwx + t % 3 - 1;
        bool pv = (sy >= 0) & (sy < H) & (sx >= 0) & (sx < W);
        size_t poff = pv ? ((size_t)((size_t)pb * H + sy) * W + sx) * Cin : (size_t)0;
        const _Float16* gsrc = Xh + poff + k0 + sub;   // safe base when invalid
        _Float16* l0 = &lb[buf][(lane >> 1) * LROW + sub];
        async_copy_b128(gsrc, l0);                     // halves [sub, sub+8)
        async_copy_b128(gsrc + 16, l0 + 16);           // halves [16+sub, 24+sub)
    };

    v8f acc = {};
    _Float16 hz = (_Float16)0.0f;
    if (wv == 0) stage(0, 0);
    int buf = 0;
    for (int idx = 0; idx < L; ++idx) {
        int t  = idx / kch;
        int k0 = (idx - t * kch) << 5;
        bool pend = (idx + 1 < L);
        if (wv == 0) {
            if (pend) stage(buf ^ 1, idx + 1);
            if (pend) async_wait_2();   // in-order completion: drains current buf
            else      async_wait_0();
        }
        __syncthreads();                // current buffer visible to all waves

        int sy = chy + t / 3 - 1;
        int sx = cwx + t % 3 - 1;
        bool valid = (sy >= 0) & (sy < H) & (sx >= 0) & (sx < W);

        const _Float16* at = arow + (size_t)t * Cin + k0;
        __builtin_prefetch(at + 32);
        v8h a0 = *(const v8h*)(at + g * 8);
        v8h a1 = *(const v8h*)(at + 16 + g * 8);
        const _Float16* lp = &lb[buf][ml * LROW + g * 8];
        v8h b0 = *(const v8h*)(lp);
        v8h b1 = *(const v8h*)(lp + 16);
        v16h af, bf;
#pragma unroll
        for (int i = 0; i < 8; ++i) {
            af[i] = a0[i]; af[i + 8] = a1[i];
            bf[i]     = valid ? b0[i] : hz;
            bf[i + 8] = valid ? b1[i] : hz;
        }
        acc = __builtin_amdgcn_wmma_f32_16x16x32_f16(false, af, false, bf,
                                                     (short)0, acc, false, false);
        __syncthreads();                // all waves done reading before restage
        buf ^= 1;
    }

#pragma unroll
    for (int v = 0; v < 8; ++v) {
        int co = co0 + v + 8 * g;
        float val = acc[v];
        if (bias) val += bias[co];
        size_t idx = (size_t)j * Cout + co;   // NHWC
        Y[idx] = val;
        if (Yh) Yh[idx] = (_Float16)val;
    }
}

// ============================ helper kernels ================================
__global__ void cast_f32_f16_kernel(const float* __restrict__ s,
                                    _Float16* __restrict__ d, long n)
{
    long i = (long)blockIdx.x * blockDim.x + threadIdx.x;
    if (i < n) d[i] = (_Float16)s[i];
}

// d[n*K + k] = s[k*N + n]   (f32 -> f16 transpose pack for wq/wk/wv)
__global__ void transpose_cast_kernel(const float* __restrict__ s,
                                      _Float16* __restrict__ d, int K, int N)
{
    long i = (long)blockIdx.x * blockDim.x + threadIdx.x;
    if (i >= (long)K * N) return;
    int n = (int)(i % N);
    int k = (int)(i / N);
    d[(size_t)n * K + k] = (_Float16)s[i];
}

// w [Cout,Cin,3,3] f32  ->  d [Cout,9,Cin] f16
__global__ void pack_w9_kernel(const float* __restrict__ w,
                               _Float16* __restrict__ d, int Cout, int Cin)
{
    long i = (long)blockIdx.x * blockDim.x + threadIdx.x;
    if (i >= (long)Cout * 9 * Cin) return;
    int ci = (int)(i % Cin);
    int t  = (int)((i / Cin) % 9);
    int co = (int)(i / ((long)9 * Cin));
    d[i] = (_Float16)w[((size_t)co * Cin + ci) * 9 + t];
}

// cat[b,h,w,c] = c<Cu ? up2(u)[...] : skip_nchw[...]; also f16 mirror
__global__ void build_cat_kernel(const float* __restrict__ u, const float* __restrict__ skip,
                                 float* __restrict__ cat, _Float16* __restrict__ cath,
                                 int B, int H, int W, int Cu, int Cs)
{
    int C = Cu + Cs;
    long total = (long)B * H * W * C;
    long i = (long)blockIdx.x * blockDim.x + threadIdx.x;
    if (i >= total) return;
    int c = (int)(i % C);
    long r = i / C;
    int wx = (int)(r % W); r /= W;
    int hy = (int)(r % H); r /= H;
    int b  = (int)r;
    float v;
    if (c < Cu)
        v = u[(((size_t)b * (H / 2) + hy / 2) * (W / 2) + wx / 2) * Cu + c];
    else
        v = skip[(((size_t)b * Cs + (c - Cu)) * H + hy) * W + wx];
    cat[i]  = v;
    cath[i] = (_Float16)v;
}

__global__ void softmax_rows_kernel(const float* __restrict__ S,
                                    _Float16* __restrict__ P, int L)
{
    long row = blockIdx.x;
    const float* s = S + (size_t)row * L;
    _Float16*    p = P + (size_t)row * L;
    __shared__ float red[256];
    int tid = threadIdx.x;
    float mx = -3.0e38f;
    for (int i = tid; i < L; i += 256) mx = fmaxf(mx, s[i]);
    red[tid] = mx; __syncthreads();
    for (int st = 128; st > 0; st >>= 1) { if (tid < st) red[tid] = fmaxf(red[tid], red[tid + st]); __syncthreads(); }
    mx = red[0]; __syncthreads();
    float sum = 0.f;
    for (int i = tid; i < L; i += 256) sum += __expf(s[i] - mx);
    red[tid] = sum; __syncthreads();
    for (int st = 128; st > 0; st >>= 1) { if (tid < st) red[tid] += red[tid + st]; __syncthreads(); }
    float inv = 1.0f / red[0];
    for (int i = tid; i < L; i += 256) p[i] = (_Float16)(__expf(s[i] - mx) * inv);
}

__global__ void layernorm_rows_kernel(float* __restrict__ X, const float* __restrict__ gm,
                                      const float* __restrict__ bt, int C)
{
    long row = blockIdx.x;
    float* x = X + (size_t)row * C;
    __shared__ float red[256];
    int tid = threadIdx.x;
    float s = 0.f;
    for (int i = tid; i < C; i += 256) s += x[i];
    red[tid] = s; __syncthreads();
    for (int st = 128; st > 0; st >>= 1) { if (tid < st) red[tid] += red[tid + st]; __syncthreads(); }
    float mu = red[0] / C; __syncthreads();
    float s2 = 0.f;
    for (int i = tid; i < C; i += 256) { float d = x[i] - mu; s2 += d * d; }
    red[tid] = s2; __syncthreads();
    for (int st = 128; st > 0; st >>= 1) { if (tid < st) red[tid] += red[tid + st]; __syncthreads(); }
    float rstd = rsqrtf(red[0] / C + 1e-5f);
    for (int i = tid; i < C; i += 256) x[i] = (x[i] - mu) * rstd * gm[i] + bt[i];
}

// cat[b] += mean_p ctx[p]  (ctx: [4, HW, C] contiguous)
__global__ void attn_add_kernel(float* __restrict__ cat, const float* __restrict__ ctx,
                                int B, long HWC)
{
    long i = (long)blockIdx.x * blockDim.x + threadIdx.x;
    if (i >= HWC) return;
    float cm = 0.25f * (ctx[i] + ctx[HWC + i] + ctx[2 * HWC + i] + ctx[3 * HWC + i]);
    for (int b = 0; b < B; ++b) cat[(size_t)b * HWC + i] += cm;
}

// per-channel mean/var over rows of X [rows, C]
__global__ void bn_stats_kernel(const float* __restrict__ X, int rows, int C,
                                float* __restrict__ mu, float* __restrict__ var)
{
    int c = blockIdx.x;
    __shared__ float rs[256], rq[256];
    int tid = threadIdx.x;
    float s = 0.f, q = 0.f;
    for (int r = tid; r < rows; r += 256) { float v = X[(size_t)r * C + c]; s += v; q += v * v; }
    rs[tid] = s; rq[tid] = q; __syncthreads();
    for (int st = 128; st > 0; st >>= 1) {
        if (tid < st) { rs[tid] += rs[tid + st]; rq[tid] += rq[tid + st]; }
        __syncthreads();
    }
    if (tid == 0) { float m = rs[0] / rows; mu[c] = m; var[c] = rq[0] / rows - m * m; }
}

__global__ void bn_relu_f16_kernel(const float* __restrict__ X, _Float16* __restrict__ Yh,
                                   const float* __restrict__ mu, const float* __restrict__ var,
                                   const float* __restrict__ gm, const float* __restrict__ bt,
                                   long total, int C)
{
    long i = (long)blockIdx.x * blockDim.x + threadIdx.x;
    if (i >= total) return;
    int c = (int)(i % C);
    float y = (X[i] - mu[c]) * rsqrtf(var[c] + 1e-5f) * gm[c] + bt[c];
    Yh[i] = (_Float16)fmaxf(y, 0.0f);
}

// h NHWC [B,H,H,C] -> out NCHW [B,C,2H,2H] nearest 2x upsample
__global__ void final_up_kernel(const float* __restrict__ h, float* __restrict__ out,
                                int B, int C, int H)
{
    long total = (long)B * C * 4 * H * H;
    long i = (long)blockIdx.x * blockDim.x + threadIdx.x;
    if (i >= total) return;
    int W2 = 2 * H;
    int x = (int)(i % W2); long r = i / W2;
    int y = (int)(r % W2); r /= W2;
    int c = (int)(r % C);  r /= C;
    int b = (int)r;
    out[i] = h[(((size_t)b * H + y / 2) * H + x / 2) * C + c];
}

// =============================== host side ==================================
static inline int cdiv(int a, int b) { return (a + b - 1) / b; }

static void launch_gemm(hipStream_t st, const _Float16* A, long lda, long sA,
                        const _Float16* Bt, long ldb, long sB,
                        float* Cf, _Float16* Ch, long ldc, long sC,
                        const float* bias, int M, int N, int K,
                        float alpha, float beta, int tstore, int batch)
{
    dim3 blk(32, 4, 1);
    dim3 grd(N / 16, cdiv(M / 16, 4), batch);
    wmma_gemm_kernel<<<grd, blk, 0, st>>>(A, lda, sA, Bt, ldb, sB,
                                          Cf, Ch, ldc, sC, bias,
                                          M, N, K, alpha, beta, tstore);
}

extern "C" void kernel_launch(void* const* d_in, const int* in_sizes, int n_in,
                              void* d_out, int out_size, void* d_ws, size_t ws_size,
                              hipStream_t stream)
{
    (void)in_sizes; (void)n_in; (void)out_size; (void)ws_size;
    const float* x     = (const float*)d_in[0];
    const float* skips[3] = { (const float*)d_in[1], (const float*)d_in[2], (const float*)d_in[3] };
    const float* w_sc  = (const float*)d_in[4];
    const float* b_sc  = (const float*)d_in[5];

    // ---- workspace carve-up (bump allocator, 256B aligned; total ~85 MB) ----
    char* base = (char*)d_ws; size_t off = 0;
    auto alloc = [&](size_t bytes) -> char* {
        char* p = base + off; off = (off + bytes + 255) & ~(size_t)255; return p;
    };
    const int B = 6;
    _Float16* xh   = (_Float16*)alloc((size_t)B * 64 * 768 * 2);   // x as NHWC f16
    _Float16* wpk  = (_Float16*)alloc((size_t)512 * 9 * 768 * 2);  // weight-pack arena (reused per stage)
    float*    hA   = (float*)   alloc((size_t)1572864 * 4);        // ping h (f32 NHWC)
    _Float16* hAh  = (_Float16*)alloc((size_t)1572864 * 2);
    float*    hB   = (float*)   alloc((size_t)1572864 * 4);        // pong h
    _Float16* hBh  = (_Float16*)alloc((size_t)1572864 * 2);
    float*    ubuf = (float*)   alloc((size_t)393216 * 4);         // 1x1-conv out (pre-upsample)
    float*    cat  = (float*)   alloc((size_t)3145728 * 4);        // concat activations f32
    _Float16* cath = (_Float16*)alloc((size_t)3145728 * 2);        // concat f16
    _Float16* qh   = (_Float16*)alloc((size_t)2097152 * 2);        // q [4,HW,C] f16
    _Float16* kh   = (_Float16*)alloc((size_t)3145728 * 2);        // k [6,HW,C] f16
    _Float16* vTh  = (_Float16*)alloc((size_t)3145728 * 2);        // v^T [C, 6*HW] f16
    float*    sbuf = (float*)   alloc((size_t)2097152 * 4);        // score chunk [4,128,HW]
    _Float16* pbuf = (_Float16*)alloc((size_t)2097152 * 2);        // softmax probs f16
    float*    ctxm = (float*)   alloc((size_t)2097152 * 4);        // ctx (mean over j) [4,HW,C]
    float*    muv  = (float*)   alloc(512 * 4);
    float*    varv = (float*)   alloc(512 * 4);

    const int T = 256;

    // x [6, 64, 768] is already NHWC [6,8,8,768]: just cast to f16
    { long n = (long)B * 64 * 768;
      cast_f32_f16_kernel<<<cdiv((int)n, T), T, 0, stream>>>(x, xh, n); }

    // stem conv: pack w_sc then 3x3 conv 768 -> 512 @ 8x8
    { long n = (long)512 * 9 * 768;
      pack_w9_kernel<<<cdiv((int)n, T), T, 0, stream>>>(w_sc, wpk, 512, 768); }
    { dim3 blk(32, 4, 1), grd((B * 8 * 8) / 16, 512 / 64, 1);
      conv3x3_wmma_kernel<<<grd, blk, 0, stream>>>(xh, wpk, b_sc, hA, hAh, 8, 8, 768, 512); }

    const int HsA[3]  = {8, 16, 32};     // spatial of block input h
    const int CinA[3] = {512, 256, 128}; // channels of block input h
    const int CupA[3] = {256, 128, 64};  // channels after 1x1 conv (== skip channels)

    float* hin = hA;  _Float16* hinh = hAh;
    float* hout = hB; _Float16* houth = hBh;

    for (int bi = 0; bi < 3; ++bi) {
        const float* w_up = (const float*)d_in[6  + 10 * bi];
        const float* b_up = (const float*)d_in[7  + 10 * bi];
        const float* wq   = (const float*)d_in[8  + 10 * bi];
        const float* wk   = (const float*)d_in[9  + 10 * bi];
        const float* wv   = (const float*)d_in[10 + 10 * bi];
        const float* lng  = (const float*)d_in[11 + 10 * bi];
        const float* lnb  = (const float*)d_in[12 + 10 * bi];
        const float* bng  = (const float*)d_in[13 + 10 * bi];
        const float* bnb  = (const float*)d_in[14 + 10 * bi];
        const float* w_cb = (const float*)d_in[15 + 10 * bi];

        int Hs = HsA[bi], Cin = CinA[bi], Cup = CupA[bi], Cs = Cup;
        int H = 2 * Hs, HW = H * H, C = Cup + Cs;
        long HWC = (long)HW * C;
        const int Tn = 128;               // divides all HW (256/1024/4096)

        _Float16* wuph = wpk;
        _Float16* wqT  = wuph + (size_t)Cup * Cin;
        _Float16* wkT  = wqT  + (size_t)C * C;
        _Float16* wvT  = wkT  + (size_t)C * C;
        _Float16* wcbp = wvT  + (size_t)C * C;

        { long n = (long)Cup * Cin; cast_f32_f16_kernel<<<cdiv((int)n, T), T, 0, stream>>>(w_up, wuph, n); }
        { long n = (long)C * C; transpose_cast_kernel<<<cdiv((int)n, T), T, 0, stream>>>(wq, wqT, C, C); }
        { long n = (long)C * C; transpose_cast_kernel<<<cdiv((int)n, T), T, 0, stream>>>(wk, wkT, C, C); }
        { long n = (long)C * C; transpose_cast_kernel<<<cdiv((int)n, T), T, 0, stream>>>(wv, wvT, C, C); }
        { long n = (long)Cup * 9 * C; pack_w9_kernel<<<cdiv((int)n, T), T, 0, stream>>>(w_cb, wcbp, Cup, C); }

        // 1x1 conv on pre-upsample h (commutes with nearest up2): u NHWC f32
        launch_gemm(stream, wuph, Cin, 0, hinh, Cin, 0, ubuf, nullptr, Cup, 0, b_up,
                    Cup, B * Hs * Hs, Cin, 1.0f, 0.0f, /*tstore=*/1, 1);

        // concat(up2(u), skip_nchw) -> cat f32 + cath f16  (NHWC == [b, hw, c])
        { long n = (long)B * HWC;
          build_cat_kernel<<<cdiv((int)n, T), T, 0, stream>>>(ubuf, skips[bi], cat, cath,
                                                              B, H, H, Cup, Cs); }

        // q (batches 1..4), k/v (all 6 batches); v stored transposed [C, 6*HW]
        launch_gemm(stream, cath + HWC, C, 0, wqT, C, 0, nullptr, qh, C, 0, nullptr,
                    4 * HW, C, C, 1.0f, 0.0f, 0, 1);
        launch_gemm(stream, cath, C, 0, wkT, C, 0, nullptr, kh, C, 0, nullptr,
                    6 * HW, C, C, 1.0f, 0.0f, 0, 1);
        launch_gemm(stream, cath, C, 0, wvT, C, 0, nullptr, vTh, (long)6 * HW, 0, nullptr,
                    6 * HW, C, C, 1.0f, 0.0f, /*tstore=*/1, 1);

        // attention, chunked over Tn query rows; j in {0,1} -> neighbor batch p + 2*j
        float inv_sqrt_c = 1.0f / sqrtf((float)C);
        for (int n0 = 0; n0 < HW; n0 += Tn) {
            for (int j = 0; j < 2; ++j) {
                int boff = j ? 2 : 0;     // bj = p + boff
                // scores[p][Tn][HW] = (q_p / sqrt(C)) @ k_{p+boff}^T
                launch_gemm(stream, qh + (size_t)n0 * C, C, (long)HW * C,
                            kh + (size_t)boff * HW * C, C, (long)HW * C,
                            sbuf, nullptr, HW, (long)Tn * HW, nullptr,
                            Tn, HW, C, inv_sqrt_c, 0.0f, 0, 4);
                softmax_rows_kernel<<<4 * Tn, 256, 0, stream>>>(sbuf, pbuf, HW);
                // ctx += 0.5 * probs @ v_{p+boff}   (mean over the 2 neighbors)
                launch_gemm(stream, pbuf, HW, (long)Tn * HW,
                            vTh + (size_t)boff * HW, (long)6 * HW, (long)HW,
                            ctxm + (size_t)n0 * C, nullptr, C, (long)HW * C, nullptr,
                            Tn, C, HW, 0.5f, j ? 1.0f : 0.0f, 0, 4);
            }
        }

        layernorm_rows_kernel<<<4 * HW, 256, 0, stream>>>(ctxm, lng, lnb, C);
        attn_add_kernel<<<cdiv((int)HWC, T), T, 0, stream>>>(cat, ctxm, B, HWC);

        // batch-norm over (B, HW) per channel, then ReLU -> f16 for the 3x3 conv
        bn_stats_kernel<<<C, 256, 0, stream>>>(cat, B * HW, C, muv, varv);
        { long n = (long)B * HWC;
          bn_relu_f16_kernel<<<cdiv((int)n, T), T, 0, stream>>>(cat, cath, muv, varv,
                                                                bng, bnb, n, C); }

        // 3x3 conv C -> Cup (no bias) -> next h (f32 + f16)
        { dim3 blkD(32, 4, 1), grd((B * HW) / 16, Cup / 64, 1);
          conv3x3_wmma_kernel<<<grd, blkD, 0, stream>>>(cath, wcbp, nullptr,
                                                        hout, houth, H, H, C, Cup); }

        // ping-pong
        float* tf = hin; hin = hout; hout = tf;
        _Float16* th = hinh; hinh = houth; houth = th;
    }

    // final 2x upsample: hin = block3 output NHWC [6,64,64,64] -> NCHW [6,64,128,128]
    { long n = (long)B * 64 * 128 * 128;
      final_up_kernel<<<cdiv((int)n, T), T, 0, stream>>>(hin, (float*)d_out, B, 64, 64); }
}
